// GraphAttentionBlock_36936718745720
// MI455X (gfx1250) — matrix operand order repaired and checked by reading, not compiled, not run on hardware
//
#include <hip/hip_runtime.h>
#include <hip/hip_bf16.h>
#include <math.h>

// ---- problem constants (match reference) ----
#define Bq   2
#define Nn   1024
#define Cc   64
#define Dd   512
#define Hh   16
#define HDd  32
#define Mm   (Nn + Cc)      // 1088
#define FF   (4 * Dd)       // 2048
#define LN_EPS 1e-5f
#define INV_SQRT_HD 0.17677669529663689f   // 1/sqrt(32)
#define INV_2LS2 (1.0f / (2.0f * 0.22f * 0.22f))

typedef __attribute__((ext_vector_type(16))) __bf16        v16bf;
typedef __attribute__((ext_vector_type(8)))  float         v8f;
typedef __attribute__((ext_vector_type(4)))  unsigned int  u32x4;

static __device__ __forceinline__ unsigned short f2bf(float f) {
    unsigned int u = __builtin_bit_cast(unsigned int, f);
    unsigned int r = u + 0x7FFFu + ((u >> 16) & 1u);   // RNE
    return (unsigned short)(r >> 16);
}

static __device__ __forceinline__ v8f vzero8() {
    v8f z;
#pragma unroll
    for (int i = 0; i < 8; ++i) z[i] = 0.0f;
    return z;
}

// Build a 16-element bf16 WMMA operand from two 16-byte chunks.
static __device__ __forceinline__ v16bf load_bf16x16(const unsigned short* p0,
                                                     const unsigned short* p1) {
    union { u32x4 q[2]; v16bf h; } u;
    u.q[0] = *reinterpret_cast<const u32x4*>(p0);
    u.q[1] = *reinterpret_cast<const u32x4*>(p1);
    return u.h;
}

// ---------------- weight cast fp32 -> bf16 ----------------
__global__ void cast_f32_bf16(const float* __restrict__ in, unsigned short* __restrict__ out, int n) {
    for (int i = blockIdx.x * blockDim.x + threadIdx.x; i < n; i += gridDim.x * blockDim.x)
        out[i] = f2bf(in[i]);
}

// ---------------- LayerNorm (row of 512) -> bf16 ----------------
__global__ __launch_bounds__(256)
void ln_kernel(const float* __restrict__ src, unsigned short* __restrict__ dst,
               const float* __restrict__ g, const float* __restrict__ beta,
               int rowsPerBatchSrc, int rowsPerBatchDst, int dstRowOffset) {
    __shared__ float s0[256], s1[256];
    const int row = blockIdx.x, tid = threadIdx.x;
    const float* x = src + (size_t)row * Dd;
    float a = x[tid], b = x[tid + 256];
    s0[tid] = a + b; s1[tid] = a * a + b * b;
    __syncthreads();
    for (int off = 128; off > 0; off >>= 1) {
        if (tid < off) { s0[tid] += s0[tid + off]; s1[tid] += s1[tid + off]; }
        __syncthreads();
    }
    float mean = s0[0] * (1.0f / Dd);
    float var  = s1[0] * (1.0f / Dd) - mean * mean;
    float inv  = rsqrtf(var + LN_EPS);
    int bb = row / rowsPerBatchSrc, i = row % rowsPerBatchSrc;
    unsigned short* d = dst + (size_t)(bb * rowsPerBatchDst + dstRowOffset + i) * Dd;
    d[tid]       = f2bf(g[tid]       * (a - mean) * inv + beta[tid]);
    d[tid + 256] = f2bf(g[tid + 256] * (b - mean) * inv + beta[tid + 256]);
}

// ---------------- generic bf16 WMMA GEMM:  out[r,o] = A[r,:] . W[o,:] + bias[o] ----------------
// Wave tile: 32 rows x 64 cols (2 A tiles reused against 4 B tiles -> 8 WMMA / k-step).
// MODE 0: outB = bf16(v*scaleOut)                (Q with 1/sqrt(hd) folded in, K)
// MODE 1: outB transposed: [(b*No + o)*ORB + m]  (V -> Vt)
// MODE 2: outF = residual + v                    (O-proj, FFN2)
// MODE 3: outB = bf16(gelu_exact(v))             (FFN1)
template <int MODE>
__global__ __launch_bounds__(128)
void gemm_bf16_wmma(const unsigned short* __restrict__ A, const unsigned short* __restrict__ W,
                    const float* __restrict__ bias, const float* __restrict__ residual,
                    float* __restrict__ outF, unsigned short* __restrict__ outB,
                    int K, int No, int ORB, int ARB, float scaleOut) {
    const int wave = threadIdx.x >> 5, lane = threadIdx.x & 31;
    const int lrow = lane & 15, half = lane >> 4;
    const int r0 = blockIdx.x * 32;
    const int cb = blockIdx.y * 256 + wave * 64;
    const int ar = (r0 / ORB) * ARB + (r0 % ORB);
    const unsigned short* ap0 = A + (size_t)(ar + lrow) * K;
    const unsigned short* ap1 = A + (size_t)(ar + 16 + lrow) * K;

    v8f acc[2][4];
#pragma unroll
    for (int g = 0; g < 2; ++g)
#pragma unroll
        for (int c = 0; c < 4; ++c) acc[g][c] = vzero8();

    for (int k = 0; k < K; k += 32) {
        if (k + 32 < K) {
            __builtin_prefetch(ap0 + k + 32, 0, 0);   // global_prefetch_b8
            __builtin_prefetch(ap1 + k + 32, 0, 0);
        }
        v16bf av0 = load_bf16x16(ap0 + k + half * 8, ap0 + k + 16 + half * 8);
        v16bf av1 = load_bf16x16(ap1 + k + half * 8, ap1 + k + 16 + half * 8);
#pragma unroll
        for (int c = 0; c < 4; ++c) {
            const unsigned short* wp = W + (size_t)(cb + c * 16 + lrow) * K + k + half * 8;
            v16bf bv = load_bf16x16(wp, wp + 16);
            acc[0][c] = __builtin_amdgcn_wmma_f32_16x16x32_bf16(false, av0, false, bv,
                                                                (short)0, acc[0][c], false, false);
            acc[1][c] = __builtin_amdgcn_wmma_f32_16x16x32_bf16(false, av1, false, bv,
                                                                (short)0, acc[1][c], false, false);
        }
    }

    const int rbase = half * 8, coll = lane & 15;
#pragma unroll
    for (int g = 0; g < 2; ++g) {
#pragma unroll
        for (int c = 0; c < 4; ++c) {
#pragma unroll
            for (int j = 0; j < 8; ++j) {
                int row = r0 + g * 16 + rbase + j;
                int col = cb + c * 16 + coll;
                float v = acc[g][c][j] + bias[col];
                if (MODE == 0) {
                    outB[(size_t)row * No + col] = f2bf(v * scaleOut);
                } else if (MODE == 1) {
                    int bb = row / ORB, m = row % ORB;
                    outB[((size_t)bb * No + col) * ORB + m] = f2bf(v);
                } else if (MODE == 2) {
                    size_t o = (size_t)row * No + col;
                    outF[o] = residual[o] + v;
                } else { // MODE 3: exact GELU
                    float ge = 0.5f * v * (1.0f + erff(v * 0.70710678118654752f));
                    outB[(size_t)row * No + col] = f2bf(ge);
                }
            }
        }
    }
}

// ---------------- fused flash attention, templated on head group ----------------
// HG 0: distance bias/scale, HG 1: adjacency, HG 2: edge conditioning, HG 3: free.
// One wave handles 16 query rows x 1 head; m processed 64 columns per step.
// grid = (N/16, 4, B), block = 32; h = HG*4 + blockIdx.y
template <int HG>
__global__ __launch_bounds__(32)
void attn_kernel(const unsigned short* __restrict__ Qb,   // [B*N, D] bf16 (pre-scaled 1/sqrt(hd))
                 const unsigned short* __restrict__ Kb,   // [B*M, D] bf16
                 const unsigned short* __restrict__ Vtb,  // [B*D, M] bf16 (transposed)
                 const float* __restrict__ positions,     // [B,N,2]
                 const float* __restrict__ adjacency,     // [B,N,N]
                 const float* __restrict__ edge,          // [B,4,N,N]
                 unsigned short* __restrict__ attB) {     // [B*N, D] bf16
    __shared__ unsigned short lds_p[16 * 64];

    const int n0 = blockIdx.x * 16;
    const int h  = HG * 4 + blockIdx.y;
    const int b  = blockIdx.z;
    const int lane = threadIdx.x;
    const int lrow = lane & 15, half = lane >> 4, rbase = half * 8;
    const int coll = lane & 15;

    // Q A-operand (held in registers for the whole m loop)
    const unsigned short* qp = Qb + (size_t)(b * Nn + n0 + lrow) * Dd + h * HDd;
    v16bf aQ = load_bf16x16(qp + half * 8, qp + 16 + half * 8);

    // per-lane query positions for distance heads (rows rbase..rbase+7)
    float qx[8], qy[8];
    if (HG == 0) {
#pragma unroll
        for (int j = 0; j < 8; ++j) {
            size_t pi = ((size_t)b * Nn + n0 + rbase + j) * 2;
            qx[j] = positions[pi];
            qy[j] = positions[pi + 1];
        }
    }

    v8f acc0 = vzero8(), acc1 = vzero8();        // 16 x 32 output accumulator (hd 0:16, 16:32)
    float mrun[8], lrun[8];
#pragma unroll
    for (int j = 0; j < 8; ++j) { mrun[j] = -INFINITY; lrun[j] = 0.0f; }

    v8f zeroc = vzero8();

    for (int m0 = 0; m0 < Mm; m0 += 64) {
        // ---- logits: S = Q . K^T  (four 16x16 WMMA tiles) ----
        v8f s[4];
#pragma unroll
        for (int t = 0; t < 4; ++t) {
            const unsigned short* kp = Kb + (size_t)(b * Mm + m0 + t * 16 + lrow) * Dd
                                          + h * HDd + half * 8;
            v16bf bK = load_bf16x16(kp, kp + 16);
            s[t] = __builtin_amdgcn_wmma_f32_16x16x32_bf16(false, aQ, false, bK,
                                                           (short)0, zeroc, false, false);
        }

        // ---- bias / scale per element (compile-time specialized by HG) ----
        float z[4][8], sc[4][8];
#pragma unroll
        for (int t = 0; t < 4; ++t) {
            const int mcol = m0 + t * 16 + coll;
            const bool inN = (mcol < Nn);
            float kx = 0.0f, ky = 0.0f;
            if (HG == 0 && inN) {
                size_t pi = ((size_t)b * Nn + mcol) * 2;
                kx = positions[pi]; ky = positions[pi + 1];
            }
#pragma unroll
            for (int j = 0; j < 8; ++j) {
                float bia, scl;
                if (HG == 0) {
                    if (inN) {
                        float dx = qx[j] - kx, dy = qy[j] - ky;
                        float dist = __expf(-(dx * dx + dy * dy) * INV_2LS2);
                        bia = 2.0f * dist - 1.0f;
                        scl = 0.25f + 0.75f * dist;
                    } else { bia = -1.0f; scl = 0.25f; }
                } else if (HG == 1) {
                    const int rn = n0 + rbase + j;
                    float a = inN ? adjacency[((size_t)b * Nn + rn) * Nn + mcol] : 0.0f;
                    bia = 2.0f * a - 1.0f;
                    scl = 0.25f + 0.75f * a;
                } else if (HG == 2) {
                    const int rn = n0 + rbase + j;
                    bia = inN ? edge[(((size_t)b * 4 + blockIdx.y) * Nn + rn) * Nn + mcol] : 0.0f;
                    scl = 1.0f;
                } else {
                    bia = 0.0f; scl = 1.0f;
                }
                z[t][j]  = s[t][j] + bia;
                sc[t][j] = scl;
            }
        }

        // ---- online softmax update (row stats across 16 lanes of each half) ----
#pragma unroll
        for (int j = 0; j < 8; ++j) {
            float mj = fmaxf(fmaxf(z[0][j], z[1][j]), fmaxf(z[2][j], z[3][j]));
#pragma unroll
            for (int msk = 1; msk < 16; msk <<= 1)
                mj = fmaxf(mj, __shfl_xor(mj, msk, 32));
            float newm = fmaxf(mrun[j], mj);
            float corr = __expf(mrun[j] - newm);
            mrun[j] = newm;
            float rs = 0.0f;
#pragma unroll
            for (int t = 0; t < 4; ++t) {
                z[t][j] = __expf(z[t][j] - newm) * sc[t][j];   // z now holds p
                rs += z[t][j];
            }
#pragma unroll
            for (int msk = 1; msk < 16; msk <<= 1)
                rs += __shfl_xor(rs, msk, 32);
            lrun[j] = lrun[j] * corr + rs;
            acc0[j] *= corr;
            acc1[j] *= corr;
        }

        // ---- transpose P through LDS: C layout -> A layout (bf16) ----
#pragma unroll
        for (int t = 0; t < 4; ++t)
#pragma unroll
            for (int j = 0; j < 8; ++j)
                lds_p[(rbase + j) * 64 + t * 16 + coll] = f2bf(z[t][j]);
        asm volatile("s_wait_dscnt 0" ::: "memory");

        // ---- accumulate: acc += P . V  (two k-chunks of 32 over m) ----
#pragma unroll
        for (int kc = 0; kc < 2; ++kc) {
            const unsigned short* lp = lds_p + lrow * 64 + kc * 32;
            v16bf aP = load_bf16x16(lp + half * 8, lp + 16 + half * 8);
            const unsigned short* vp0 = Vtb + ((size_t)b * Dd + h * HDd + lrow) * Mm
                                            + m0 + kc * 32 + half * 8;
            v16bf bV0 = load_bf16x16(vp0, vp0 + 16);
            acc0 = __builtin_amdgcn_wmma_f32_16x16x32_bf16(false, aP, false, bV0,
                                                           (short)0, acc0, false, false);
            const unsigned short* vp1 = Vtb + ((size_t)b * Dd + h * HDd + 16 + lrow) * Mm
                                            + m0 + kc * 32 + half * 8;
            v16bf bV1 = load_bf16x16(vp1, vp1 + 16);
            acc1 = __builtin_amdgcn_wmma_f32_16x16x32_bf16(false, aP, false, bV1,
                                                           (short)0, acc1, false, false);
        }
    }

    // ---- epilogue: out = acc / max(l, 1e-6) ----
#pragma unroll
    for (int j = 0; j < 8; ++j) {
        float rl = 1.0f / fmaxf(lrun[j], 1e-6f);
        size_t o = (size_t)(b * Nn + n0 + rbase + j) * Dd + h * HDd + coll;
        attB[o]      = f2bf(acc0[j] * rl);
        attB[o + 16] = f2bf(acc1[j] * rl);
    }
}

// ---------------- host-side launch ----------------
extern "C" void kernel_launch(void* const* d_in, const int* in_sizes, int n_in,
                              void* d_out, int out_size, void* d_ws, size_t ws_size,
                              hipStream_t stream) {
    (void)in_sizes; (void)n_in; (void)out_size; (void)ws_size;
    const float* hidden    = (const float*)d_in[0];
    const float* adjacency = (const float*)d_in[1];
    const float* positions = (const float*)d_in[2];
    const float* context   = (const float*)d_in[3];
    const float* edge      = (const float*)d_in[4];
    const float* Wq = (const float*)d_in[5];  const float* bq = (const float*)d_in[6];
    const float* Wk = (const float*)d_in[7];  const float* bk = (const float*)d_in[8];
    const float* Wv = (const float*)d_in[9];  const float* bv = (const float*)d_in[10];
    const float* Wo = (const float*)d_in[11]; const float* bo = (const float*)d_in[12];
    const float* ln1g = (const float*)d_in[13]; const float* ln1b = (const float*)d_in[14];
    const float* ln2g = (const float*)d_in[15]; const float* ln2b = (const float*)d_in[16];
    const float* W1 = (const float*)d_in[17]; const float* b1 = (const float*)d_in[18];
    const float* W2 = (const float*)d_in[19]; const float* b2 = (const float*)d_in[20];
    float* out = (float*)d_out;

    // workspace carve-up (bytes, 256-aligned sizes)
    char* ws = (char*)d_ws;
    size_t o = 0;
    auto take = [&](size_t bytes) { char* p = ws + o; o += (bytes + 255) & ~(size_t)255; return p; };
    unsigned short* WqB = (unsigned short*)take((size_t)Dd * Dd * 2);
    unsigned short* WkB = (unsigned short*)take((size_t)Dd * Dd * 2);
    unsigned short* WvB = (unsigned short*)take((size_t)Dd * Dd * 2);
    unsigned short* WoB = (unsigned short*)take((size_t)Dd * Dd * 2);
    unsigned short* W1B = (unsigned short*)take((size_t)FF * Dd * 2);
    unsigned short* W2B = (unsigned short*)take((size_t)Dd * FF * 2);
    unsigned short* ksB = (unsigned short*)take((size_t)Bq * Mm * Dd * 2);
    unsigned short* QB  = (unsigned short*)take((size_t)Bq * Nn * Dd * 2);
    unsigned short* KB  = (unsigned short*)take((size_t)Bq * Mm * Dd * 2);
    unsigned short* VtB = (unsigned short*)take((size_t)Bq * Dd * Mm * 2);
    unsigned short* atB = (unsigned short*)take((size_t)Bq * Nn * Dd * 2);
    float*          h2F = (float*)take((size_t)Bq * Nn * Dd * 4);
    unsigned short* n2B = (unsigned short*)take((size_t)Bq * Nn * Dd * 2);
    unsigned short* gB  = (unsigned short*)take((size_t)Bq * Nn * FF * 2);

    // 1) cast weights to bf16
    cast_f32_bf16<<<1024, 256, 0, stream>>>(Wq, WqB, Dd * Dd);
    cast_f32_bf16<<<1024, 256, 0, stream>>>(Wk, WkB, Dd * Dd);
    cast_f32_bf16<<<1024, 256, 0, stream>>>(Wv, WvB, Dd * Dd);
    cast_f32_bf16<<<1024, 256, 0, stream>>>(Wo, WoB, Dd * Dd);
    cast_f32_bf16<<<2048, 256, 0, stream>>>(W1, W1B, FF * Dd);
    cast_f32_bf16<<<2048, 256, 0, stream>>>(W2, W2B, Dd * FF);

    // 2) LN1: hidden rows and context rows into key_source [B, M, D] bf16
    ln_kernel<<<Bq * Nn, 256, 0, stream>>>(hidden,  ksB, ln1g, ln1b, Nn, Mm, 0);
    ln_kernel<<<Bq * Cc, 256, 0, stream>>>(context, ksB, ln1g, ln1b, Cc, Mm, Nn);

    // 3) projections (bf16 WMMA). Q pre-scaled by 1/sqrt(hd); V written transposed.
    gemm_bf16_wmma<0><<<dim3(Bq * Nn / 32, Dd / 256), 128, 0, stream>>>(
        ksB, WqB, bq, nullptr, nullptr, QB,  Dd, Dd, Nn, Mm, INV_SQRT_HD);
    gemm_bf16_wmma<0><<<dim3(Bq * Mm / 32, Dd / 256), 128, 0, stream>>>(
        ksB, WkB, bk, nullptr, nullptr, KB,  Dd, Dd, Mm, Mm, 1.0f);
    gemm_bf16_wmma<1><<<dim3(Bq * Mm / 32, Dd / 256), 128, 0, stream>>>(
        ksB, WvB, bv, nullptr, nullptr, VtB, Dd, Dd, Mm, Mm, 1.0f);

    // 4) fused flash attention, one launch per head group (compile-time bias path)
    attn_kernel<0><<<dim3(Nn / 16, 4, Bq), 32, 0, stream>>>(QB, KB, VtB, positions, adjacency, edge, atB);
    attn_kernel<1><<<dim3(Nn / 16, 4, Bq), 32, 0, stream>>>(QB, KB, VtB, positions, adjacency, edge, atB);
    attn_kernel<2><<<dim3(Nn / 16, 4, Bq), 32, 0, stream>>>(QB, KB, VtB, positions, adjacency, edge, atB);
    attn_kernel<3><<<dim3(Nn / 16, 4, Bq), 32, 0, stream>>>(QB, KB, VtB, positions, adjacency, edge, atB);

    // 5) output projection + residual -> hidden2 (fp32)
    gemm_bf16_wmma<2><<<dim3(Bq * Nn / 32, Dd / 256), 128, 0, stream>>>(
        atB, WoB, bo, hidden, h2F, nullptr, Dd, Dd, Nn, Nn, 1.0f);

    // 6) LN2 -> n2 bf16
    ln_kernel<<<Bq * Nn, 256, 0, stream>>>(h2F, n2B, ln2g, ln2b, Nn, Nn, 0);

    // 7) FFN1 (+ exact GELU) -> bf16
    gemm_bf16_wmma<3><<<dim3(Bq * Nn / 32, FF / 256), 128, 0, stream>>>(
        n2B, W1B, b1, nullptr, nullptr, gB, Dd, FF, Nn, Nn, 1.0f);

    // 8) FFN2 + residual -> final output (fp32)
    gemm_bf16_wmma<2><<<dim3(Bq * Nn / 32, Dd / 256), 128, 0, stream>>>(
        gB, W2B, b2, h2F, out, nullptr, FF, Dd, Nn, Nn, 1.0f);
}
